// SeperateLoss_59433757442322
// MI455X (gfx1250) — matrix operand order repaired and checked by reading, not compiled, not run on hardware
//
#include <hip/hip_runtime.h>

#define NROWS 16384
#define DDIM  256

typedef __attribute__((ext_vector_type(16))) __bf16 v16bf;
typedef __attribute__((ext_vector_type(8)))  float  v8f;
typedef __attribute__((ext_vector_type(4)))  __bf16 v4bf;

union FragB { v16bf v; uint4 q[2]; };

// ---------------------------------------------------------------------------
// Zero the scalar output (harness poisons d_out; atomics below need 0 start).
// ---------------------------------------------------------------------------
__global__ void zero_out_kernel(float* out) {
  if (threadIdx.x == 0) out[0] = 0.0f;
}

// ---------------------------------------------------------------------------
// Split fp32 -> bf16 hi + bf16 lo residual (Markidis-style precision split).
// One thread handles 4 consecutive elements (float4 in, 2x v4bf out).
// ---------------------------------------------------------------------------
__global__ void split_bf16_kernel(const float* __restrict__ x,
                                  __bf16* __restrict__ xh,
                                  __bf16* __restrict__ xl) {
  const int i = blockIdx.x * blockDim.x + threadIdx.x;   // float4 index
  const float4 f = ((const float4*)x)[i];
  __bf16 h0 = (__bf16)f.x, h1 = (__bf16)f.y, h2 = (__bf16)f.z, h3 = (__bf16)f.w;
  __bf16 l0 = (__bf16)(f.x - (float)h0);
  __bf16 l1 = (__bf16)(f.y - (float)h1);
  __bf16 l2 = (__bf16)(f.z - (float)h2);
  __bf16 l3 = (__bf16)(f.w - (float)h3);
  v4bf vh = {h0, h1, h2, h3};
  v4bf vl = {l0, l1, l2, l3};
  ((v4bf*)xh)[i] = vh;
  ((v4bf*)xl)[i] = vl;
}

// ---------------------------------------------------------------------------
// Fused Gram x x^T -> exp -> row-sum -> log -> atomic scalar.
// 256 workgroups (8 wave32 each); one workgroup owns 64 rows, sweeps all cols.
//   wave w: wm = w>>1 (0..3) -> 16-row tile; wn = w&1 (0..1) -> 32-col slab.
// A (16 rows x 256 K, hi+lo) is held ENTIRELY in VGPRs per wave (zero steady-
// state A traffic). B fragments are identical across the 4 wm-waves of a wn
// group -> served from WGP$ after the first wave touches them (64KB/j-tile).
// K accumulated over 8 WMMA steps of K=32 with the 3-term bf16 split
// (hi*hi + hi*lo + lo*hi) for ~fp32-accurate dot products.
// ---------------------------------------------------------------------------
__global__ __launch_bounds__(256)
void gram_lse_kernel(const __bf16* __restrict__ xh,
                     const __bf16* __restrict__ xl,
                     float* __restrict__ out) {
  __shared__ float rowsum[64];
  __shared__ float partial[2];

  const int tid  = threadIdx.x;
  const int lane = tid & 31;
  const int w    = tid >> 5;
  const int wm   = w >> 1;        // 0..3 : 16-row tile within the 64-row block
  const int wn   = w & 1;         // 0..1 : 32-col slab within the 64-col tile
  const int half = lane >> 4;     // lane group (K-interleave per ISA layout)
  const int l16  = lane & 15;
  const int rowBlock = blockIdx.x * 64;

  if (tid < 64) rowsum[tid] = 0.0f;

  // Load this wave's A tile (16 rows x 256 K, hi+lo) into registers.
  // A 16x32 bf16 layout per k-step: lanes 0-15 hold K=[kb..kb+7, kb+16..23],
  // lanes 16-31 hold K=[kb+8..15, kb+24..31] -> two 16B chunks per fragment.
  const uint4* gAh = (const uint4*)xh;
  const uint4* gAl = (const uint4*)xl;
  const size_t aBase = (size_t)(rowBlock + wm * 16 + l16) * (DDIM / 8);
  FragB Ah[8], Al[8];
#pragma unroll
  for (int k = 0; k < 8; ++k) {
    Ah[k].q[0] = gAh[aBase + k * 4 + half];
    Ah[k].q[1] = gAh[aBase + k * 4 + half + 2];
    Al[k].q[0] = gAl[aBase + k * 4 + half];
    Al[k].q[1] = gAl[aBase + k * 4 + half + 2];
  }

  float acc[8] = {0.f, 0.f, 0.f, 0.f, 0.f, 0.f, 0.f, 0.f};

  for (int j0 = 0; j0 < NROWS; j0 += 64) {
    const int col0 = j0 + wn * 32 + l16;
    const int col1 = col0 + 16;
    const uint4* b0h = (const uint4*)(xh + (size_t)col0 * DDIM);
    const uint4* b0l = (const uint4*)(xl + (size_t)col0 * DDIM);
    const uint4* b1h = (const uint4*)(xh + (size_t)col1 * DDIM);
    const uint4* b1l = (const uint4*)(xl + (size_t)col1 * DDIM);

    // Prefetch next j-tile's B rows (global_prefetch_b8) while we compute.
    if (j0 + 64 < NROWS) {
      __builtin_prefetch(xh + (size_t)(col0 + 64) * DDIM, 0, 0);
      __builtin_prefetch(xl + (size_t)(col0 + 64) * DDIM, 0, 0);
      __builtin_prefetch(xh + (size_t)(col1 + 64) * DDIM, 0, 0);
      __builtin_prefetch(xl + (size_t)(col1 + 64) * DDIM, 0, 0);
    }

    v8f c0 = {0.f, 0.f, 0.f, 0.f, 0.f, 0.f, 0.f, 0.f};
    v8f c1 = {0.f, 0.f, 0.f, 0.f, 0.f, 0.f, 0.f, 0.f};

#pragma unroll
    for (int k = 0; k < 8; ++k) {
      FragB f0h, f0l, f1h, f1l;
      // B 32x16 layout: lane's column row, K = [kb+half*8..+7, kb+16+half*8..+7].
      const int bo = k * 4 + half;
      f0h.q[0] = b0h[bo]; f0h.q[1] = b0h[bo + 2];
      f0l.q[0] = b0l[bo]; f0l.q[1] = b0l[bo + 2];
      f1h.q[0] = b1h[bo]; f1h.q[1] = b1h[bo + 2];
      f1l.q[0] = b1l[bo]; f1l.q[1] = b1l[bo + 2];

      c0 = __builtin_amdgcn_wmma_f32_16x16x32_bf16(false, Ah[k].v, false, f0h.v, (short)0, c0, false, false);
      c1 = __builtin_amdgcn_wmma_f32_16x16x32_bf16(false, Ah[k].v, false, f1h.v, (short)0, c1, false, false);
      c0 = __builtin_amdgcn_wmma_f32_16x16x32_bf16(false, Ah[k].v, false, f0l.v, (short)0, c0, false, false);
      c1 = __builtin_amdgcn_wmma_f32_16x16x32_bf16(false, Ah[k].v, false, f1l.v, (short)0, c1, false, false);
      c0 = __builtin_amdgcn_wmma_f32_16x16x32_bf16(false, Al[k].v, false, f0h.v, (short)0, c0, false, false);
      c1 = __builtin_amdgcn_wmma_f32_16x16x32_bf16(false, Al[k].v, false, f1h.v, (short)0, c1, false, false);
    }

    // C layout: VGPR r, lane<16 -> row r; lane>=16 -> row r+8. Column index is
    // irrelevant for the row-sum, so accumulate both tiles into acc[r].
#pragma unroll
    for (int r = 0; r < 8; ++r)
      acc[r] += __expf(2.0f * c0[r]) + __expf(2.0f * c1[r]);
  }

  __syncthreads();   // rowsum[] zeroed before any atomic lands

  // In-wave reduction over the 16 lanes of each half, then one LDS atomic per
  // row per wave (only the 2 wn-waves of a row tile collide).
#pragma unroll
  for (int r = 0; r < 8; ++r) {
    float v = acc[r];
    v += __shfl_xor(v, 1, 32);
    v += __shfl_xor(v, 2, 32);
    v += __shfl_xor(v, 4, 32);
    v += __shfl_xor(v, 8, 32);
    if (l16 == 0) unsafeAtomicAdd(&rowsum[wm * 16 + half * 8 + r], v);
  }
  __syncthreads();

  if (tid < 64) {
    float v = __logf(rowsum[tid]);
#pragma unroll
    for (int off = 16; off; off >>= 1) v += __shfl_down(v, off, 32);
    if ((tid & 31) == 0) partial[tid >> 5] = v;
  }
  __syncthreads();
  if (tid == 0) unsafeAtomicAdd(out, (partial[0] + partial[1]) * 0.01f);  // 1/C
}

// ---------------------------------------------------------------------------
extern "C" void kernel_launch(void* const* d_in, const int* in_sizes, int n_in,
                              void* d_out, int out_size, void* d_ws, size_t ws_size,
                              hipStream_t stream) {
  (void)in_sizes; (void)n_in; (void)out_size; (void)ws_size;
  const float* x = (const float*)d_in[0];
  float* out = (float*)d_out;
  __bf16* xh = (__bf16*)d_ws;                       // 8 MB
  __bf16* xl = xh + (size_t)NROWS * DDIM;           // 8 MB

  zero_out_kernel<<<1, 32, 0, stream>>>(out);
  split_bf16_kernel<<<(NROWS * DDIM / 4) / 256, 256, 0, stream>>>(x, xh, xl);
  gram_lse_kernel<<<NROWS / 64, 256, 0, stream>>>(xh, xl, out);
}
